// Net_37778532336274
// MI455X (gfx1250) — matrix operand order repaired and checked by reading, not compiled, not run on hardware
//
#include <hip/hip_runtime.h>
#include <hip/hip_bf16.h>

// ---------------------------------------------------------------------------
// CDNA5 (gfx1250, wave32) bf16 WMMA GEMM pipeline, software-pipelined:
// prologue loads k=0 fragments; each loop body issues ALL next-k loads
// (one clause), then runs NT v_wmma_f32_16x16x32_bf16 on current fragments,
// then rotates registers. Epilogue iteration consumes the final fragments.
// Operands padded/transposed in workspace so the hot loop has NO guards:
//   A fragment (16x32, row-major [M x K]): lane m = m0+(lane&15),
//     two contiguous 16B loads at koff=k0+8*(lane>>4) and koff+16.
//   B fragment from TRANSPOSED B (Bt[N][K]): lane n = n0+(lane&15),
//     two contiguous 16B loads at koff=k0+8*(lane>>4) and koff+16.
//   C/D: VGPR j -> M = m0 + j + 8*(lane>>4), N = n0 + (lane&15).
// ---------------------------------------------------------------------------

typedef __bf16 v16bf __attribute__((ext_vector_type(16)));
typedef __bf16 v8bf  __attribute__((ext_vector_type(8)));
typedef float  v8f   __attribute__((ext_vector_type(8)));
typedef float  v8fl  __attribute__((ext_vector_type(8)));

__device__ inline v16bf load_a_bf16(const __bf16* __restrict__ A, int lda,
                                    long long m, int koff) {
    const __bf16* p = A + m * lda + koff;
    const v8bf lo = *(const v8bf*)p;          // K = koff .. koff+7
    const v8bf hi = *(const v8bf*)(p + 16);   // K = koff+16 .. koff+23
    v16bf a;
#pragma unroll
    for (int j = 0; j < 8; ++j) { a[j] = lo[j]; a[8 + j] = hi[j]; }
    return a;
}

__device__ inline v16bf load_a_f32(const float* __restrict__ A, int lda,
                                   long long m, int koff) {
    const float* p = A + m * lda + koff;
    const v8fl lo = *(const v8fl*)p;
    const v8fl hi = *(const v8fl*)(p + 16);
    v16bf a;
#pragma unroll
    for (int j = 0; j < 8; ++j) { a[j] = (__bf16)lo[j]; a[8 + j] = (__bf16)hi[j]; }
    return a;
}

__device__ inline v16bf load_bt(const __bf16* __restrict__ Bt, int ldk,
                                int n, int koff) {
    const __bf16* p = Bt + (long long)n * ldk + koff;
    const v8bf lo = *(const v8bf*)p;          // K = koff .. koff+7
    const v8bf hi = *(const v8bf*)(p + 16);   // K = koff+16 .. koff+23
    v16bf b;
#pragma unroll
    for (int j = 0; j < 8; ++j) { b[j] = lo[j]; b[8 + j] = hi[j]; }
    return b;
}

// emode 0: off = batch*strideO + m*ldo + n        (store if m < mstore)
// emode 1: b2 = m/divm, r = m%divm; off = b2*strideO + n*ldo + r   (transposed)
// act: 0=none 1=ELU 2=ReLU 3=sigmoid
template <bool A_F32, int NT>
__global__ void gemm_wmma_kernel(const void* __restrict__ Aptr,
                                 const __bf16* __restrict__ Btp,
                                 const float* __restrict__ bias,
                                 void* __restrict__ Outptr,
                                 int K, int lda, int ldk,
                                 int mtiles, int ngroups,
                                 long long strideA, long long strideB,
                                 int emode, int divm, long long strideO,
                                 int ldo, int mstore, int nstore,
                                 int act, int out_bf16) {
    const int wave = threadIdx.x >> 5;
    const int lane = threadIdx.x & 31;
    const int tile = blockIdx.x * (blockDim.x >> 5) + wave;
    if (tile >= mtiles * ngroups) return;   // whole-wave exit, EXEC stays full
    const int mt = tile / ngroups;
    const int ng = tile % ngroups;
    const int half = lane >> 4;
    const int n_lane = lane & 15;
    const long long m = (long long)mt * 16 + n_lane;   // A row for this lane
    const long long batch = blockIdx.y;

    const float*  Af = A_F32 ? ((const float*)Aptr) + batch * strideA : nullptr;
    const __bf16* Ab = A_F32 ? nullptr : ((const __bf16*)Aptr) + batch * strideA;
    const __bf16* Bt = Btp + batch * strideB;

    v8f acc[NT] = {};
    v16bf a_cur, b_cur[NT];

    // ---- pipeline prologue: fragments for k0 = 0 ----
    {
        const int koffA = (half << 3);
        if constexpr (A_F32) a_cur = load_a_f32(Af, lda, m, koffA);
        else                 a_cur = load_a_bf16(Ab, lda, m, koffA);
        const int koffB = (half << 3);
#pragma unroll
        for (int t = 0; t < NT; ++t)
            b_cur[t] = load_bt(Bt, ldk, (ng * NT + t) * 16 + n_lane, koffB);
    }

    // ---- steady state: issue k0+32 loads, then NT WMMAs on k0, rotate ----
    for (int k0 = 0; k0 + 32 < K; k0 += 32) {
        const int kn = k0 + 32;
        const int koffA = kn + (half << 3);
        v16bf a_nxt, b_nxt[NT];
        if constexpr (A_F32) {
            a_nxt = load_a_f32(Af, lda, m, koffA);
            __builtin_prefetch(Af + m * lda + koffA + 64, 0, 3);
        } else {
            a_nxt = load_a_bf16(Ab, lda, m, koffA);
            __builtin_prefetch(Ab + m * lda + koffA + 64, 0, 3);
        }
        const int koffB = kn + (half << 3);
#pragma unroll
        for (int t = 0; t < NT; ++t)
            b_nxt[t] = load_bt(Bt, ldk, (ng * NT + t) * 16 + n_lane, koffB);

#pragma unroll
        for (int t = 0; t < NT; ++t)
            acc[t] = __builtin_amdgcn_wmma_f32_16x16x32_bf16(
                false, a_cur, false, b_cur[t], (short)0, acc[t], false, false);

        a_cur = a_nxt;
#pragma unroll
        for (int t = 0; t < NT; ++t) b_cur[t] = b_nxt[t];
    }

    // ---- pipeline epilogue: last k-step ----
#pragma unroll
    for (int t = 0; t < NT; ++t)
        acc[t] = __builtin_amdgcn_wmma_f32_16x16x32_bf16(
            false, a_cur, false, b_cur[t], (short)0, acc[t], false, false);

    // ---- bias + activation + store ----
#pragma unroll
    for (int t = 0; t < NT; ++t) {
        const int n = (ng * NT + t) * 16 + n_lane;
        const float bval = (bias != nullptr && n < nstore) ? bias[n] : 0.0f;
#pragma unroll
        for (int j = 0; j < 8; ++j) {
            const long long mrow = (long long)mt * 16 + j + (half << 3);
            float v = acc[t][j] + bval;
            if (act == 1)      v = (v > 0.0f) ? v : (__expf(v) - 1.0f);
            else if (act == 2) v = fmaxf(v, 0.0f);
            else if (act == 3) v = 1.0f / (1.0f + __expf(-v));
            bool ok = (n < nstore);
            long long offo;
            if (emode == 0) {
                ok = ok && (mrow < (long long)mstore);
                offo = batch * strideO + mrow * (long long)ldo + n;
            } else {
                const long long b2 = mrow / divm;
                const long long r  = mrow - b2 * (long long)divm;
                offo = b2 * strideO + (long long)n * ldo + r;
            }
            if (ok) {
                if (out_bf16) ((__bf16*)Outptr)[offo] = (__bf16)v;
                else          ((float*)Outptr)[offo]  = v;
            }
        }
    }
}

// dst[Rp][Cp] bf16; transpose=0: dst[r][c]=src[r][c] (r<S0,c<S1) else 0
//             transpose=1: dst[r][c]=src[c][r] (c<S0,r<S1) else 0
__global__ void cvt_pad_kernel(const float* __restrict__ src,
                               __bf16* __restrict__ dst,
                               int S0, int S1, int Rp, int Cp, int transpose) {
    const long long i = (long long)blockIdx.x * blockDim.x + threadIdx.x;
    if (i >= (long long)Rp * Cp) return;
    const int r = (int)(i / Cp), c = (int)(i % Cp);
    float v = 0.0f;
    if (transpose) { if (c < S0 && r < S1) v = src[(long long)c * S1 + r]; }
    else           { if (r < S0 && c < S1) v = src[(long long)r * S1 + c]; }
    dst[i] = (__bf16)v;
}

// buf[row][ld]: zero columns [start, start+len) for all rows
__global__ void zero_tail_kernel(__bf16* __restrict__ buf, int ld,
                                 int start, int len, long long rows) {
    const long long i = (long long)blockIdx.x * blockDim.x + threadIdx.x;
    if (i >= rows * len) return;
    const long long row = i / len;
    const int k = (int)(i % len);
    buf[row * ld + start + k] = (__bf16)0.0f;
}

// ---------------------------------------------------------------------------

extern "C" void kernel_launch(void* const* d_in, const int* in_sizes, int n_in,
                              void* d_out, int out_size, void* d_ws, size_t ws_size,
                              hipStream_t stream) {
    (void)in_sizes; (void)n_in; (void)out_size; (void)ws_size;
    constexpr int B = 8192, N = 200, F_IN = 64, CH = 32, H1 = 512, H2 = 256;
    constexpr int MP = 208;          // node dim padded for M tiles (13*16)
    constexpr int KP = 224;          // node dim padded for K steps (7*32)
    constexpr long long TSTRIDE = (long long)CH * KP;   // per-batch t0t/t1t

    const float* x   = (const float*)d_in[0];
    const float* adj = (const float*)d_in[1];
    const float* W1  = (const float*)d_in[2];
    const float* b1  = (const float*)d_in[3];
    const float* W2  = (const float*)d_in[4];
    const float* b2  = (const float*)d_in[5];
    const float* Wf1 = (const float*)d_in[6];
    const float* bf1 = (const float*)d_in[7];
    const float* Wf2 = (const float*)d_in[8];
    const float* bf2 = (const float*)d_in[9];
    const float* Wf3 = (const float*)d_in[10];
    const float* bf3 = (const float*)d_in[11];

    char* ws = (char*)d_ws;
    size_t off = 0;
    auto alloc = [&](size_t bytes) -> void* {
        off = (off + 255) & ~(size_t)255;
        void* p = ws + off; off += bytes; return p;
    };

    __bf16* a_bf  = (__bf16*)alloc((size_t)MP * KP * 2);        // [208][224]
    __bf16* w1t   = (__bf16*)alloc((size_t)CH * F_IN * 2);      // [32][64]
    __bf16* w2t   = (__bf16*)alloc((size_t)CH * CH * 2);        // [32][32]
    __bf16* wf1t  = (__bf16*)alloc((size_t)H1 * (N * CH) * 2);  // [512][6400]
    __bf16* wf2t  = (__bf16*)alloc((size_t)H2 * H1 * 2);        // [256][512]
    __bf16* wf3t  = (__bf16*)alloc((size_t)16 * H2 * 2);        // [16][256] pad
    __bf16* t0t   = (__bf16*)alloc((size_t)B * CH * KP * 2);    // [B][32][224]
    __bf16* h1    = (__bf16*)alloc((size_t)B * MP * CH * 2);    // [B][208][32]
    __bf16* h2    = h1;                                          // [B][6400]
    __bf16* h3    = (__bf16*)alloc((size_t)B * H1 * 2);
    __bf16* h4    = (__bf16*)alloc((size_t)B * H2 * 2);

    auto cvt = [&](const float* s, __bf16* d, int S0, int S1, int Rp, int Cp, int tr) {
        const long long n = (long long)Rp * Cp;
        cvt_pad_kernel<<<(unsigned)((n + 255) / 256), 256, 0, stream>>>(s, d, S0, S1, Rp, Cp, tr);
    };
    cvt(adj, a_bf, N, N, MP, KP, 0);
    cvt(W1,  w1t,  F_IN, CH, CH, F_IN, 1);
    cvt(W2,  w2t,  CH, CH, CH, CH, 1);
    cvt(Wf1, wf1t, N * CH, H1, H1, N * CH, 1);
    cvt(Wf2, wf2t, H1, H2, H2, H1, 1);
    cvt(Wf3, wf3t, H2, 1, 16, H2, 1);

    {   // zero node-pad region [200,224) of t0t for all (b, ch) rows
        const long long rows = (long long)B * CH, len = KP - N;
        zero_tail_kernel<<<(unsigned)((rows * len + 255) / 256), 256, 0, stream>>>(
            t0t, KP, N, (int)len, rows);
    }

    const dim3 blk(256);
    auto blocks = [](long long tiles) { return (unsigned)((tiles + 7) / 8); };

    // 1) t0t[b][c][node] = (x @ W1)^T : A f32 [B*200, 64], NT=2, emode1 div=200
    {
        const int mtiles = (B * N) / 16;   // 102400, all full
        gemm_wmma_kernel<true, 2><<<dim3(blocks(mtiles), 1), blk, 0, stream>>>(
            x, w1t, nullptr, t0t, F_IN, F_IN, F_IN, mtiles, 1, 0, 0,
            /*emode*/1, /*divm*/N, TSTRIDE, /*ldo*/KP, 0, CH, /*act*/0, 1);
    }
    // 2) h1[b][m][c] = ELU(adj @ t0 + b1) : per-sample [208x224]@[224x32]
    {
        const int mtiles = MP / 16;        // 13
        gemm_wmma_kernel<false, 2><<<dim3(blocks(mtiles), B), blk, 0, stream>>>(
            a_bf, t0t, b1, h1, KP, KP, KP, mtiles, 1, 0, TSTRIDE,
            0, 0, (long long)MP * CH, CH, MP, CH, /*ELU*/1, 1);
    }
    // 3) t1t[b][c][node] = (h1 @ W2)^T : A bf16 [B*208, 32], emode1 div=208
    {
        const int mtiles = (B * MP) / 16;  // 106496
        gemm_wmma_kernel<false, 2><<<dim3(blocks(mtiles), 1), blk, 0, stream>>>(
            h1, w2t, nullptr, t0t, CH, CH, CH, mtiles, 1, 0, 0,
            1, MP, TSTRIDE, KP, 0, CH, 0, 1);
    }
    // 4) h2[b][node*32+c] = ELU(adj @ t1 + b2), store only node<200
    {
        const int mtiles = MP / 16;
        gemm_wmma_kernel<false, 2><<<dim3(blocks(mtiles), B), blk, 0, stream>>>(
            a_bf, t0t, b2, h2, KP, KP, KP, mtiles, 1, 0, TSTRIDE,
            0, 0, (long long)N * CH, CH, /*mstore*/N, CH, 1, 1);
    }
    // 5) h3 = relu(h2 @ Wf1 + bf1) : [8192,6400]@[6400,512], NT=4
    {
        const int mtiles = B / 16, ngroups = (H1 / 16) / 4;   // 512, 8
        gemm_wmma_kernel<false, 4><<<dim3(blocks((long long)mtiles * ngroups), 1), blk, 0, stream>>>(
            h2, wf1t, bf1, h3, N * CH, N * CH, N * CH, mtiles, ngroups, 0, 0,
            0, 0, 0, H1, B, H1, /*ReLU*/2, 1);
    }
    // 6) h4 = relu(h3 @ Wf2 + bf2) : [8192,512]@[512,256], NT=4
    {
        const int mtiles = B / 16, ngroups = (H2 / 16) / 4;   // 512, 4
        gemm_wmma_kernel<false, 4><<<dim3(blocks((long long)mtiles * ngroups), 1), blk, 0, stream>>>(
            h3, wf2t, bf2, h4, H1, H1, H1, mtiles, ngroups, 0, 0,
            0, 0, 0, H2, B, H2, 2, 1);
    }
    // 7) out = sigmoid(h4 @ Wf3 + bf3) : [8192,256]@[256,1] -> f32, NT=1
    {
        const int mtiles = B / 16;         // 512
        gemm_wmma_kernel<false, 1><<<dim3(blocks(mtiles), 1), blk, 0, stream>>>(
            h4, wf3t, bf3, d_out, H2, H2, H2, mtiles, 1, 0, 0,
            0, 0, 0, 1, B, /*nstore*/1, /*sigmoid*/3, 0);
    }
}
